// Net_24489903522726
// MI455X (gfx1250) — matrix operand order repaired and checked by reading, not compiled, not run on hardware
//
#include <hip/hip_runtime.h>

typedef _Float16 v16h __attribute__((ext_vector_type(16)));
typedef _Float16 v8h  __attribute__((ext_vector_type(8)));
typedef float    v8f  __attribute__((ext_vector_type(8)));

// ---------------------------------------------------------------------------
// D = A*B + C, A/B f16 16x32 / 32x16, C/D f32 16x16 (CDNA5 WMMA, wave32)
// ---------------------------------------------------------------------------
__device__ __forceinline__ v8f wmma_f16(v16h a, v16h b, v8f c) {
  return __builtin_amdgcn_wmma_f32_16x16x32_f16(
      /*neg_a=*/false, a, /*neg_b=*/false, b,
      /*c_mod=*/(short)0, c, /*reuse_a=*/false, /*reuse_b=*/false);
}

__device__ __forceinline__ v16h cat8(v8h lo, v8h hi) {
  return __builtin_shufflevector(lo, hi, 0,1,2,3,4,5,6,7,8,9,10,11,12,13,14,15);
}

// A-fragment (16-bit A 16x32 layout): lane = h*16+m holds row M=m,
// element j -> K = kc*32 + (j/8)*16 + h*8 + (j%8)  => two contiguous 8-half runs
__device__ __forceinline__ v16h load_a_frag(const _Float16* rowbase, int kc, int h) {
  const v8h* p0 = (const v8h*)(rowbase + kc * 32 + h * 8);
  const v8h* p1 = (const v8h*)(rowbase + kc * 32 + 16 + h * 8);
  return cat8(*p0, *p1);
}

// B-fragment: pre-swizzled in workspace, lane's 16 halves contiguous (32B)
__device__ __forceinline__ v16h load_b_frag(const _Float16* p) {
  const v8h* q = (const v8h*)p;
  return cat8(q[0], q[1]);
}

// ---------------------------------------------------------------------------
// Stage a [K,64] f32 matrix into f16 B-fragments (16-bit B 32x16 layout).
// Chunk order (nt, p); for cores (perm=1): p = kc*64 + r, K-tile = r*4 + kc.
// Within a chunk: lane l (h=l/16, n=l%16), element j -> K = k32*32 + (j/8)*16 + h*8 + (j%8),
// column = nt*16 + n.
// ---------------------------------------------------------------------------
__global__ void stage_b(const float* __restrict__ src, _Float16* __restrict__ dst,
                        int kTiles, int permCore, int total) {
  int t = blockIdx.x * blockDim.x + threadIdx.x;
  if (t >= total) return;
  int lane = t & 31;
  int p    = (t >> 5) % kTiles;
  int nt   = (t >> 5) / kTiles;
  int k32  = permCore ? ((p & 63) * 4 + (p >> 6)) : p;
  int h = lane >> 4, n = lane & 15;
  int col = nt * 16 + n;
  long kb = (long)k32 * 32 + h * 8;
#pragma unroll
  for (int j = 0; j < 16; ++j) {
    dst[(long)t * 16 + j] = (_Float16)src[(kb + (j & 7) + (j >> 3) * 16) * 64 + col];
  }
}

// ---------------------------------------------------------------------------
// Fused LayerNorm + tensor-train chain. Block = 128 threads (4 waves),
// covering 64 batch rows (4 M-tiles) and all 64 output columns (4 N-tiles).
// K-split: wave wv owns kc-slice wv of K for ALL 4 M-tiles, so waves read
// disjoint quarters of the staged cores (4x less L2 traffic). Partial C
// tiles are reduced across waves through LDS (f16 partials, f32 sum).
// ---------------------------------------------------------------------------
#define XST 136   // xnh d-stride (halves): 128 + 8 pad, 16B-aligned runs
#define TST 72    // t / pacc row stride (halves): 64 + 8 pad

__device__ __forceinline__ void reduce_pacc(const _Float16* pacc, _Float16* tT,
                                            int tid, bool transpose) {
  // 64x64 elements, 4 partials each; [r][M] layout in pacc blocks of 64*TST
#pragma unroll 1
  for (int idx = tid; idx < 4096; idx += 128) {
    int M = idx & 63, r = idx >> 6;
    float s = (float)pacc[0 * 64 * TST + r * TST + M] +
              (float)pacc[1 * 64 * TST + r * TST + M] +
              (float)pacc[2 * 64 * TST + r * TST + M] +
              (float)pacc[3 * 64 * TST + r * TST + M];
    if (transpose) tT[M * TST + r] = (_Float16)s;   // [M][r] for final GEMM A
    else           tT[r * TST + M] = (_Float16)s;   // [r][M] for t-broadcast
  }
}

__global__ __launch_bounds__(128) void tt_fused(
    const float* __restrict__ x, const float* __restrict__ lnw,
    const float* __restrict__ lnb,
    const _Float16* __restrict__ bL0, const _Float16* __restrict__ bC1,
    const _Float16* __restrict__ bC2, const _Float16* __restrict__ bLast,
    float* __restrict__ out) {
  __shared__ __align__(16) _Float16 xnh[3][64][XST];   // normalized acts, f16
  __shared__ __align__(16) _Float16 pacc[4][64][TST];  // per-wave partial C
  __shared__ __align__(16) _Float16 tT[64 * TST];      // reduced t tile
  __shared__ float muA[64];
  __shared__ float rsA[64];

  const int tid  = threadIdx.x;
  const int lane = tid & 31;
  const int wv   = tid >> 5;
  const int m    = lane & 15;
  const int h    = lane >> 4;
  const long base = (long)blockIdx.x * 64 * 384;

  // ---- pass 1: per-batch mean / rstd over the 384 [D,L] values ----
  if (tid < 64) {
    const float4* p = (const float4*)(x + base + tid * 384);
    float s = 0.f, ss = 0.f;
#pragma unroll 4
    for (int i = 0; i < 96; ++i) {
      float4 v = p[i];
      s  += v.x + v.y + v.z + v.w;
      ss += v.x * v.x + v.y * v.y + v.z * v.z + v.w * v.w;
    }
    float mu  = s * (1.0f / 384.0f);
    float var = ss * (1.0f / 384.0f) - mu * mu;
    muA[tid] = mu;
    rsA[tid] = rsqrtf(var + 1e-5f);
  }
  __syncthreads();

  // ---- pass 2: normalize + affine, scatter f16 as [l][b][d] ----
  for (int v4 = tid; v4 < 64 * 96; v4 += 128) {
    int b  = v4 / 96;
    int r4 = (v4 - b * 96) * 4;  // element offset within batch, = d*3+l
    float4 xv = *(const float4*)(x + base + b * 384 + r4);
    float4 w4 = *(const float4*)(lnw + r4);
    float4 b4 = *(const float4*)(lnb + r4);
    float mu = muA[b], rs = rsA[b];
    float xa[4] = {xv.x, xv.y, xv.z, xv.w};
    float wa[4] = {w4.x, w4.y, w4.z, w4.w};
    float ba[4] = {b4.x, b4.y, b4.z, b4.w};
#pragma unroll
    for (int e = 0; e < 4; ++e) {
      int dl = r4 + e;
      xnh[dl % 3][b][dl / 3] = (_Float16)((xa[e] - mu) * rs * wa[e] + ba[e]);
    }
  }
  __syncthreads();

  // ---- step 0: t0 = xn[:,:,0] @ layer0 (K=128, kc-slice wv per wave) ----
  {
    v8f acc[4][4] = {};  // [mt][nt]
    v16h a[4];
#pragma unroll
    for (int mt = 0; mt < 4; ++mt)
      a[mt] = load_a_frag(&xnh[0][mt * 16 + m][0], wv, h);
#pragma unroll
    for (int nt = 0; nt < 4; ++nt) {
      v16h b = load_b_frag(bL0 + ((nt * 4 + wv) * 32 + lane) * 16);
#pragma unroll
      for (int mt = 0; mt < 4; ++mt) acc[mt][nt] = wmma_f16(a[mt], b, acc[mt][nt]);
    }
#pragma unroll
    for (int mt = 0; mt < 4; ++mt)
#pragma unroll
      for (int nt = 0; nt < 4; ++nt)
#pragma unroll
        for (int i = 0; i < 8; ++i)
          pacc[wv][nt * 16 + m][mt * 16 + h * 8 + i] = (_Float16)acc[mt][nt][i];
  }
  __syncthreads();
  reduce_pacc(&pacc[0][0][0], tT, tid, false);
  __syncthreads();

  // ---- steps 1 & 2: Khatri-Rao GEMMs, K=8192, kc-slice wv per wave ----
#pragma unroll 1
  for (int step = 0; step < 2; ++step) {
    const _Float16* bC = step ? bC2 : bC1;
    v8f acc[4][4] = {};  // [mt][nt]
    v16h va[4];
#pragma unroll
    for (int mt = 0; mt < 4; ++mt)
      va[mt] = load_a_frag(&xnh[1 + step][mt * 16 + m][0], wv, h);
    const _Float16* bkc = bC + wv * 64 * 512 + lane * 16;
#pragma unroll 1
    for (int r = 0; r < 64; ++r) {
      // A chunks: z = t[b, r] * xn[b, d]  (8x v_pk_mul_f16 per M-tile)
      v16h a[4];
#pragma unroll
      for (int mt = 0; mt < 4; ++mt) {
        _Float16 tv = tT[r * TST + mt * 16 + m];
        v16h ts;
#pragma unroll
        for (int j = 0; j < 16; ++j) ts[j] = tv;
        a[mt] = va[mt] * ts;
      }
#pragma unroll
      for (int nt = 0; nt < 4; ++nt) {
        v16h b = load_b_frag(bkc + nt * 131072 + r * 512);
#pragma unroll
        for (int mt = 0; mt < 4; ++mt)
          acc[mt][nt] = wmma_f16(a[mt], b, acc[mt][nt]);
      }
    }
    __syncthreads();  // previous tT fully consumed; pacc free for reuse
#pragma unroll
    for (int mt = 0; mt < 4; ++mt)
#pragma unroll
      for (int nt = 0; nt < 4; ++nt)
#pragma unroll
        for (int i = 0; i < 8; ++i)
          pacc[wv][nt * 16 + m][mt * 16 + h * 8 + i] = (_Float16)acc[mt][nt][i];
    __syncthreads();
    // step 1 -> t1 as [r][M]; step 2 -> t2 transposed [M][r] (A of last GEMM)
    reduce_pacc(&pacc[0][0][0], tT, tid, step == 1);
    __syncthreads();
  }

  // ---- step 3: out = t2 @ last (K=64), wave wv owns M-tile wv ----
  {
    v8f acc[4] = {};
    const _Float16* arow = &tT[(wv * 16 + m) * TST];
#pragma unroll
    for (int kc = 0; kc < 2; ++kc) {
      v16h a = load_a_frag(arow, kc, h);
#pragma unroll
      for (int nt = 0; nt < 4; ++nt) {
        v16h b = load_b_frag(bLast + ((nt * 2 + kc) * 32 + lane) * 16);
        acc[nt] = wmma_f16(a, b, acc[nt]);
      }
    }
    float* op = out + ((long)blockIdx.x * 64 + wv * 16 + h * 8) * 64;
#pragma unroll
    for (int nt = 0; nt < 4; ++nt)
#pragma unroll
      for (int i = 0; i < 8; ++i)
        op[i * 64 + nt * 16 + m] = acc[nt][i];
  }
}

// ---------------------------------------------------------------------------
extern "C" void kernel_launch(void* const* d_in, const int* in_sizes, int n_in,
                              void* d_out, int out_size, void* d_ws, size_t ws_size,
                              hipStream_t stream) {
  const float* x      = (const float*)d_in[0];
  const float* layer0 = (const float*)d_in[1];
  const float* core1  = (const float*)d_in[2];
  const float* core2  = (const float*)d_in[3];
  const float* last   = (const float*)d_in[4];
  const float* lnw    = (const float*)d_in[5];
  const float* lnb    = (const float*)d_in[6];

  const int B = in_sizes[0] / 384;  // 32768
  _Float16* ws  = (_Float16*)d_ws;
  _Float16* c1  = ws;                       // 524288 halves (1 MB)
  _Float16* c2  = ws + 524288;              // 524288 halves
  _Float16* l0  = ws + 1048576;             // 8192 halves
  _Float16* lst = ws + 1048576 + 8192;      // 4096 halves

  // convert + swizzle weights into f16 B-fragment layout (L2-resident)
  stage_b<<<128, 256, 0, stream>>>(core1, c1, 256, 1, 32768);
  stage_b<<<128, 256, 0, stream>>>(core2, c2, 256, 1, 32768);
  stage_b<<<2,   256, 0, stream>>>(layer0, l0, 4, 0, 512);
  stage_b<<<1,   256, 0, stream>>>(last,  lst, 2, 0, 256);

  tt_fused<<<B / 64, 128, 0, stream>>>(x, lnw, lnb, l0, c1, c2, lst,
                                       (float*)d_out);
}
